// baselineGAT_35553739276823
// MI455X (gfx1250) — matrix operand order repaired and checked by reading, not compiled, not run on hardware
//
#include <hip/hip_runtime.h>
#include <hip/hip_bf16.h>
#include <hip/hip_fp16.h>

#define N_NODES 50000
#define N_EDGES 800000
#define TOT_E   (N_EDGES + N_NODES)
#define HC      128
#define CH      32
#define NH      4
#define NL      3
#define NG      128
#define NOUT    10
#define BN_EPS  1e-5f

typedef __attribute__((ext_vector_type(16))) _Float16 v16h;
typedef __attribute__((ext_vector_type(8)))  _Float16 v8h;
typedef __attribute__((ext_vector_type(4)))  _Float16 v4h;
typedef __attribute__((ext_vector_type(8)))  float    v8f;
typedef __attribute__((ext_vector_type(4)))  float    v4f;

typedef __attribute__((address_space(3))) _Float16 lds_f16_t;

// ---------------------------------------------------------------- embedding
__global__ void k_embed(const int* __restrict__ xidx,
                        const float* __restrict__ embed,
                        float* __restrict__ x) {
  int i = blockIdx.x * 256 + threadIdx.x;
  if (i >= N_NODES * HC) return;
  int n = i >> 7, j = i & 127;
  x[i] = embed[xidx[n] * HC + j];
}

// ---------------------------------------------------------------- batchnorm
__global__ void k_zero_stats(float* __restrict__ s) { s[threadIdx.x] = 0.f; }

__global__ void k_bn_stats(const float* __restrict__ x, float* __restrict__ stats) {
  __shared__ float ssum[256], ssq[256];
  int tid  = threadIdx.x;
  int f    = tid & 127, half = tid >> 7;
  int r0   = blockIdx.x * 512 + half;
  int rEnd = blockIdx.x * 512 + 512;
  if (rEnd > N_NODES) rEnd = N_NODES;
  float s = 0.f, q = 0.f;
  for (int r = r0; r < rEnd; r += 2) {
    float v = x[r * HC + f];
    s += v; q += v * v;
  }
  ssum[tid] = s; ssq[tid] = q;
  __syncthreads();
  if (half == 0) {
    s += ssum[tid + 128]; q += ssq[tid + 128];
    atomicAdd(&stats[f], s);
    atomicAdd(&stats[128 + f], q);
  }
}

// Fold BN into the linear layer: Weff[c,f] = W[c,f]*a[f] (f16),
// beff[c] = lin_b[c] + sum_f bfold[f]*W[c,f]. Also cvt gat_W to f16.
__global__ void k_bn_finalize(const float* __restrict__ stats,
                              const float* __restrict__ gamma,
                              const float* __restrict__ beta,
                              const float* __restrict__ linW,
                              const float* __restrict__ linb,
                              const float* __restrict__ gatW,
                              _Float16* __restrict__ Weff,
                              float* __restrict__ beff,
                              _Float16* __restrict__ gatWh, int l) {
  __shared__ float aS[HC], bS[HC];
  int t = threadIdx.x;
  if (t < HC) {
    float mean = stats[t] * (1.f / (float)N_NODES);
    float var  = stats[HC + t] * (1.f / (float)N_NODES) - mean * mean;
    float a    = gamma[l * HC + t] * rsqrtf(var + BN_EPS);
    aS[t] = a;
    bS[t] = beta[l * HC + t] - mean * a;
  }
  __syncthreads();
  const float* W = linW + l * CH * HC;
  for (int i = t; i < CH * HC; i += 256) {
    Weff[i]  = (_Float16)(W[i] * aS[i & 127]);
    gatWh[i] = (_Float16)(gatW[l * HC * CH + i]);
  }
  if (t < CH) {
    float s = linb[l * CH + t];
    for (int f = 0; f < HC; ++f) s += bS[f] * W[t * HC + f];
    beff[t] = s;
  }
}

// ------------------------------------------------- fused node-wise GEMMs
// per 128-row block: y = relu(x @ Weff^T + beff)  [WMMA, K=128 -> 32]
//                    h = y @ gatW^T               [WMMA, K=32  -> 128]
//                    a_src/a_dst head dots; h -> global f16 via async DMA
__global__ void __launch_bounds__(256)
k_node_gemms(const float* __restrict__ xin,
             const _Float16* __restrict__ Weff,
             const float* __restrict__ beff,
             const _Float16* __restrict__ gatWh,
             const float* __restrict__ attSrc,
             const float* __restrict__ attDst,
             _Float16* __restrict__ hbuf,
             float* __restrict__ asrc,
             float* __restrict__ adst, int l) {
  __shared__ __align__(32) _Float16 ldsXH[128 * 128];  // X (stage1), H (stage2+)
  __shared__ __align__(32) _Float16 ldsY[128 * 32];
  __shared__ float attSl[HC], attDl[HC];

  const int tid     = threadIdx.x;
  const int rowBase = blockIdx.x * 128;

  if (tid < HC) {
    attSl[tid] = attSrc[l * HC + tid];
    attDl[tid] = attDst[l * HC + tid];
  }
  // stage X tile into LDS as f16 (coalesced float4 reads)
  for (int j = tid; j < 128 * 32; j += 256) {
    int r = j >> 5, c4 = (j & 31) << 2;
    int gr = rowBase + r;
    v4f v = {0.f, 0.f, 0.f, 0.f};
    if (gr < N_NODES) v = *(const v4f*)(xin + gr * HC + c4);
    ldsXH[r * 128 + c4 + 0] = (_Float16)v[0];
    ldsXH[r * 128 + c4 + 1] = (_Float16)v[1];
    ldsXH[r * 128 + c4 + 2] = (_Float16)v[2];
    ldsXH[r * 128 + c4 + 3] = (_Float16)v[3];
  }
  __syncthreads();

  const int lane = tid & 31, wid = tid >> 5;
  const int laneLo = lane & 15, sel = lane >> 4;
  const int wrow = wid * 16;  // 16-row tile per wave

  // ---- stage 1: y = relu(x @ Weff^T + beff), 16 rows x 32 cols
  // A fragments (one per K-step) hoisted: loaded once, reused for both n-tiles.
  v16h afrag[4];
#pragma unroll
  for (int kt = 0; kt < 4; ++kt) {
    const _Float16* ap = ldsXH + (wrow + laneLo) * 128 + kt * 32 + sel * 8;
    v8h alo = *(const v8h*)ap;
    v8h ahi = *(const v8h*)(ap + 16);
#pragma unroll
    for (int i = 0; i < 8; ++i) { afrag[kt][i] = alo[i]; afrag[kt][i + 8] = ahi[i]; }
  }
#pragma unroll
  for (int ntile = 0; ntile < 2; ++ntile) {
    v8f acc = {0.f, 0.f, 0.f, 0.f, 0.f, 0.f, 0.f, 0.f};
#pragma unroll
    for (int kt = 0; kt < 4; ++kt) {
      const _Float16* bp = Weff + (ntile * 16 + laneLo) * 128 + kt * 32 + sel * 8;
      v8h blo = *(const v8h*)bp;
      v8h bhi = *(const v8h*)(bp + 16);
      v16h b;
#pragma unroll
      for (int i = 0; i < 8; ++i) { b[i] = blo[i]; b[i + 8] = bhi[i]; }
      acc = __builtin_amdgcn_wmma_f32_16x16x32_f16(false, afrag[kt], false, b,
                                                   (short)0, acc, false, false);
    }
    float bias = beff[ntile * 16 + laneLo];
#pragma unroll
    for (int d = 0; d < 8; ++d) {
      float v = acc[d] + bias;
      v = v > 0.f ? v : 0.f;
      ldsY[(wrow + sel * 8 + d) * 32 + ntile * 16 + laneLo] = (_Float16)v;
    }
  }
  __syncthreads();  // all X reads done; ldsXH reusable for H

  // ---- stage 2: h = y @ gatW^T, 16 rows x 128 cols (K=32, 1 WMMA/tile)
  {
    const _Float16* ap = ldsY + (wrow + laneLo) * 32 + sel * 8;
    v8h alo = *(const v8h*)ap;
    v8h ahi = *(const v8h*)(ap + 16);
    v16h a;
#pragma unroll
    for (int i = 0; i < 8; ++i) { a[i] = alo[i]; a[i + 8] = ahi[i]; }
#pragma unroll
    for (int ntile = 0; ntile < 8; ++ntile) {
      v8f acc = {0.f, 0.f, 0.f, 0.f, 0.f, 0.f, 0.f, 0.f};
      const _Float16* bp = gatWh + (ntile * 16 + laneLo) * 32 + sel * 8;
      v8h blo = *(const v8h*)bp;
      v8h bhi = *(const v8h*)(bp + 16);
      v16h b;
#pragma unroll
      for (int i = 0; i < 8; ++i) { b[i] = blo[i]; b[i + 8] = bhi[i]; }
      acc = __builtin_amdgcn_wmma_f32_16x16x32_f16(false, a, false, b,
                                                   (short)0, acc, false, false);
#pragma unroll
      for (int d = 0; d < 8; ++d)
        ldsXH[(wrow + sel * 8 + d) * 128 + ntile * 16 + laneLo] = (_Float16)acc[d];
    }
  }
  __syncthreads();

  // ---- stage 3a: per-head attention dots a_src/a_dst
  if (tid < 128) {
    int gr = rowBase + tid;
    if (gr < N_NODES) {
      float as[NH] = {0.f, 0.f, 0.f, 0.f};
      float ad[NH] = {0.f, 0.f, 0.f, 0.f};
#pragma unroll
      for (int c = 0; c < HC; ++c) {
        float v = (float)ldsXH[tid * 128 + c];
        as[c >> 5] += v * attSl[c];
        ad[c >> 5] += v * attDl[c];
      }
#pragma unroll
      for (int h = 0; h < NH; ++h) {
        asrc[gr * NH + h] = as[h];
        adst[gr * NH + h] = ad[h];
      }
    }
  }
  // ---- stage 3b: H tile -> global (f16) via async LDS->memory DMA.
  // Each lane issues one 16B transfer; EXEC masks out-of-range rows.
  for (int j = tid; j < 128 * 32; j += 256) {
    int r = j >> 5, gr = rowBase + r;
    if (gr < N_NODES) {
      const _Float16* gptr = hbuf + (size_t)gr * HC + (j & 31) * 4;
      unsigned lds_off = (unsigned)(uintptr_t)(lds_f16_t*)(ldsXH + j * 4);
      asm volatile("global_store_async_from_lds_b128 %0, %1, off"
                   :: "v"(gptr), "v"(lds_off) : "memory");
    }
  }
  // drain async DMA before LDS is deallocated at kernel end
  asm volatile("s_wait_asynccnt 0x0" ::: "memory");
}

// ------------------------------------------------- attention scatter phase
__global__ void k_init_agg(float* __restrict__ xout, float* __restrict__ z,
                           const float* __restrict__ gatb, int l) {
  int i = blockIdx.x * 256 + threadIdx.x;
  if (i >= N_NODES * HC) return;
  int j = i & 127;
  xout[i] = gatb[l * HC + j];
  if (j < NH) z[(i >> 7) * NH + j] = 0.f;
}

__global__ void k_edge_denom(const int* __restrict__ ei,
                             const float* __restrict__ asrc,
                             const float* __restrict__ adst,
                             float* __restrict__ z) {
  int i = blockIdx.x * 256 + threadIdx.x;
  if (i >= TOT_E * NH) return;
  int e = i >> 2, h = i & 3;
  int s, d;
  if (e < N_EDGES) { s = ei[e]; d = ei[N_EDGES + e]; } else { s = d = e - N_EDGES; }
  float v = asrc[s * NH + h] + adst[d * NH + h];
  v = v > 0.f ? v : 0.2f * v;            // leaky_relu(0.2)
  atomicAdd(&z[d * NH + h], __expf(v));
}

// one wave per edge: lane handles 4 channels of one head
__global__ void k_edge_aggregate(const int* __restrict__ ei,
                                 const float* __restrict__ asrc,
                                 const float* __restrict__ adst,
                                 const float* __restrict__ z,
                                 const _Float16* __restrict__ hbuf,
                                 float* __restrict__ xout) {
  int lane = threadIdx.x & 31;
  int e = blockIdx.x * 8 + (threadIdx.x >> 5);
  if (e >= TOT_E) return;
  int s, d;
  if (e < N_EDGES) { s = ei[e]; d = ei[N_EDGES + e]; } else { s = d = e - N_EDGES; }
  int h = lane >> 3;
  float v = asrc[s * NH + h] + adst[d * NH + h];
  v = v > 0.f ? v : 0.2f * v;
  float alpha = __expf(v) / (z[d * NH + h] + 1e-16f);
  int c0 = lane * 4;
  v4h hv = *(const v4h*)(hbuf + s * HC + c0);
#pragma unroll
  for (int i = 0; i < 4; ++i)
    atomicAdd(&xout[d * HC + c0 + i], (float)hv[i] * alpha);
}

// ---------------------------------------------------------------- readout
__global__ void k_zero_out(float* __restrict__ out) {
  int i = blockIdx.x * 256 + threadIdx.x;
  if (i < NG * NOUT) out[i] = 0.f;
}

__global__ void k_readout(const float* __restrict__ x,
                          const float* __restrict__ W,
                          const float* __restrict__ b,
                          const int* __restrict__ batch,
                          float* __restrict__ out) {
  int i = blockIdx.x * 256 + threadIdx.x;
  if (i >= N_NODES * NOUT) return;
  int n = i / NOUT, o = i - n * NOUT;
  const float* xr = x + n * HC;
  const float* wr = W + o * HC;
  float s = b[o];
#pragma unroll 4
  for (int k = 0; k < HC; k += 4) {
    v4f xv = *(const v4f*)(xr + k);
    v4f wv = *(const v4f*)(wr + k);
    s += xv[0] * wv[0] + xv[1] * wv[1] + xv[2] * wv[2] + xv[3] * wv[3];
  }
  atomicAdd(&out[batch[n] * NOUT + o], s);
}

// ---------------------------------------------------------------- driver
extern "C" void kernel_launch(void* const* d_in, const int* in_sizes, int n_in,
                              void* d_out, int out_size, void* d_ws, size_t ws_size,
                              hipStream_t stream) {
  (void)in_sizes; (void)n_in; (void)out_size; (void)ws_size;
  const int*   x_idx = (const int*)d_in[0];
  const int*   ei    = (const int*)d_in[1];
  const int*   batch = (const int*)d_in[2];
  const float* embed = (const float*)d_in[3];
  const float* gamma = (const float*)d_in[4];
  const float* beta  = (const float*)d_in[5];
  const float* linW  = (const float*)d_in[6];
  const float* linb  = (const float*)d_in[7];
  const float* gatW  = (const float*)d_in[8];
  const float* attS  = (const float*)d_in[9];
  const float* attD  = (const float*)d_in[10];
  const float* gatb  = (const float*)d_in[11];
  const float* roW   = (const float*)d_in[12];
  const float* rob   = (const float*)d_in[13];
  float* out = (float*)d_out;

  char* ws = (char*)d_ws;
  size_t off = 0;
  auto alloc = [&](size_t bytes) -> void* {
    void* p = ws + off;
    off += (bytes + 255) & ~(size_t)255;
    return p;
  };
  float*    xf    = (float*)alloc((size_t)N_NODES * HC * 4);   // features (in-place agg)
  _Float16* hbuf  = (_Float16*)alloc((size_t)N_NODES * HC * 2);
  float*    asrc  = (float*)alloc((size_t)N_NODES * NH * 4);
  float*    adst  = (float*)alloc((size_t)N_NODES * NH * 4);
  float*    z     = (float*)alloc((size_t)N_NODES * NH * 4);
  float*    stats = (float*)alloc(256 * 4);
  _Float16* Weff  = (_Float16*)alloc((size_t)CH * HC * 2);
  float*    beff  = (float*)alloc(CH * 4);
  _Float16* gatWh = (_Float16*)alloc((size_t)HC * CH * 2);

  k_embed<<<(N_NODES * HC + 255) / 256, 256, 0, stream>>>(x_idx, embed, xf);
  for (int l = 0; l < NL; ++l) {
    k_zero_stats<<<1, 256, 0, stream>>>(stats);
    k_bn_stats<<<(N_NODES + 511) / 512, 256, 0, stream>>>(xf, stats);
    k_bn_finalize<<<1, 256, 0, stream>>>(stats, gamma, beta, linW, linb, gatW,
                                         Weff, beff, gatWh, l);
    k_node_gemms<<<(N_NODES + 127) / 128, 256, 0, stream>>>(
        xf, Weff, beff, gatWh, attS, attD, hbuf, asrc, adst, l);
    k_init_agg<<<(N_NODES * HC + 255) / 256, 256, 0, stream>>>(xf, z, gatb, l);
    k_edge_denom<<<(TOT_E * NH + 255) / 256, 256, 0, stream>>>(ei, asrc, adst, z);
    k_edge_aggregate<<<(TOT_E + 7) / 8, 256, 0, stream>>>(ei, asrc, adst, z, hbuf, xf);
  }
  k_zero_out<<<(NG * NOUT + 255) / 256, 256, 0, stream>>>(out);
  k_readout<<<(N_NODES * NOUT + 255) / 256, 256, 0, stream>>>(xf, roW, rob, batch, out);
}
